// PointPillarsEncoder_86947317940413
// MI455X (gfx1250) — matrix operand order repaired and checked by reading, not compiled, not run on hardware
//
#include <hip/hip_runtime.h>

typedef unsigned short u16t;
typedef __bf16 v16bf  __attribute__((ext_vector_type(16)));
typedef __bf16 bf16x8 __attribute__((ext_vector_type(8)));
typedef float  v8f    __attribute__((ext_vector_type(8)));

#define NXc 176
#define NYc 200
#define NZc 10
#define NVc 352000
#define NBATCH 2
#define NPOINTS 100000

__device__ __forceinline__ u16t f2bf(float f) {
  unsigned u = __builtin_bit_cast(unsigned, f);
  u += 0x7FFFu + ((u >> 16) & 1u);           // round-to-nearest-even
  return (u16t)(u >> 16);
}
__device__ __forceinline__ float bf2f(u16t h) {
  unsigned u = ((unsigned)h) << 16;
  return __builtin_bit_cast(float, u);
}
__device__ __forceinline__ bf16x8 bzero8() {
  bf16x8 r;
#pragma unroll
  for (int i = 0; i < 8; ++i) r[i] = (__bf16)0.0f;
  return r;
}
__device__ __forceinline__ v8f fzero8() {
  v8f r;
#pragma unroll
  for (int i = 0; i < 8; ++i) r[i] = 0.0f;
  return r;
}

// ---------------------------------------------------------------------------
// 1) last-write-wins scatter: atomicMax of global point order per voxel
// ---------------------------------------------------------------------------
__global__ __launch_bounds__(256) void scatter_kernel(const float* __restrict__ pts,
                                                      int* __restrict__ last) {
  int i = blockIdx.x * blockDim.x + threadIdx.x;
  if (i >= NBATCH * NPOINTS) return;
  int b = i / NPOINTS;
  float x = pts[3 * (size_t)i + 0];
  float y = pts[3 * (size_t)i + 1];
  float z = pts[3 * (size_t)i + 2];
  int ix = (int)((x - 0.0f) / 0.4f);    // trunc toward zero like astype(int32)
  int iy = (int)((y + 40.0f) / 0.4f);
  int iz = (int)((z + 3.0f) / 0.4f);
  ix = min(max(ix, 0), NXc - 1);
  iy = min(max(iy, 0), NYc - 1);
  iz = min(max(iz, 0), NZc - 1);
  int flat = (iz * NYc + iy) * NXc + ix;
  atomicMax(last + (size_t)b * NVc + flat, i);
}

// ---------------------------------------------------------------------------
// 2a) fold FE batchnorms into weights/bias
// ---------------------------------------------------------------------------
__global__ void prep_fe_kernel(const float* __restrict__ w1, const float* __restrict__ b1,
                               const float* __restrict__ g1, const float* __restrict__ bb1,
                               const float* __restrict__ m1, const float* __restrict__ v1,
                               const float* __restrict__ w2, const float* __restrict__ b2,
                               const float* __restrict__ g2, const float* __restrict__ bb2,
                               const float* __restrict__ m2, const float* __restrict__ v2,
                               float* __restrict__ fw1, float* __restrict__ fb1,
                               float* __restrict__ fw2, float* __restrict__ fb2) {
  int t = threadIdx.x;
  if (t < 32) {
    float s = g1[t] / sqrtf(v1[t] + 1e-5f);
    fb1[t] = b1[t] * s + bb1[t] - m1[t] * s;
#pragma unroll
    for (int c = 0; c < 3; ++c) fw1[t * 3 + c] = w1[t * 3 + c] * s;
  }
  if (t < 64) {
    float s = g2[t] / sqrtf(v2[t] + 1e-5f);
    fb2[t] = b2[t] * s + bb2[t] - m2[t] * s;
    for (int i = 0; i < 32; ++i) fw2[t * 32 + i] = w2[t * 32 + i] * s;
  }
}

// ---------------------------------------------------------------------------
// 2b) conv weight prep: OIHW f32 -> [OC][(kh*KS+kw)*IC+ic] bf16 with BN scale
//     folded in; bias' = conv_b*s + bn_b - m*s
// ---------------------------------------------------------------------------
__global__ __launch_bounds__(256) void prep_conv_kernel(
    const float* __restrict__ w, const float* __restrict__ cb,
    const float* __restrict__ g, const float* __restrict__ bb,
    const float* __restrict__ m, const float* __restrict__ v,
    u16t* __restrict__ wq, float* __restrict__ bq, int OC, int IC, int KS) {
  int K = KS * KS * IC;
  int idx = blockIdx.x * blockDim.x + threadIdx.x;
  if (idx >= OC * K) return;
  int oc = idx / K;
  int k = idx - oc * K;
  int grp = k / IC;
  int ic = k - grp * IC;
  int kh = grp / KS;
  int kw = grp - kh * KS;
  float s = g[oc] / sqrtf(v[oc] + 1e-5f);
  float wv = w[(((size_t)oc * IC + ic) * KS + kh) * KS + kw];
  wq[(size_t)oc * K + k] = f2bf(wv * s);
  if (k == 0) bq[oc] = cb[oc] * s + bb[oc] - m[oc] * s;
}

// ---------------------------------------------------------------------------
// 3) FE pointwise MLP 3->32->64, BN folded, ReLU; writes bf16 NHWC pseudo-image
//    (NHWC flat index == voxel*64 + c since H*W == NV)
// ---------------------------------------------------------------------------
__global__ __launch_bounds__(256) void fe_kernel(const int* __restrict__ last,
                                                 const float* __restrict__ pts,
                                                 const float* __restrict__ fw1,
                                                 const float* __restrict__ fb1,
                                                 const float* __restrict__ fw2,
                                                 const float* __restrict__ fb2,
                                                 u16t* __restrict__ img) {
  __shared__ float sw1[96], sb1[32], sw2[2048], sb2[64];
  for (int i = threadIdx.x; i < 2048; i += blockDim.x) sw2[i] = fw2[i];
  if (threadIdx.x < 96) sw1[threadIdx.x] = fw1[threadIdx.x];
  if (threadIdx.x < 32) sb1[threadIdx.x] = fb1[threadIdx.x];
  if (threadIdx.x < 64) sb2[threadIdx.x] = fb2[threadIdx.x];
  __syncthreads();

  int vtx = blockIdx.x * blockDim.x + threadIdx.x;
  if (vtx >= NBATCH * NVc) return;
  int li = last[vtx];
  float p0 = 0.f, p1 = 0.f, p2 = 0.f;
  if (li >= 0) {
    p0 = pts[3 * (size_t)li + 0];
    p1 = pts[3 * (size_t)li + 1];
    p2 = pts[3 * (size_t)li + 2];
  }
  float h[32];
#pragma unroll
  for (int o = 0; o < 32; ++o) {
    float a = sw1[o * 3] * p0 + sw1[o * 3 + 1] * p1 + sw1[o * 3 + 2] * p2 + sb1[o];
    h[o] = a > 0.f ? a : 0.f;
  }
  u16t* op = img + (size_t)vtx * 64;
  for (int c8 = 0; c8 < 8; ++c8) {
    union { u16t s[8]; uint4 v4; } u;
#pragma unroll
    for (int j = 0; j < 8; ++j) {
      int c = c8 * 8 + j;
      float a = sb2[c];
#pragma unroll
      for (int i = 0; i < 32; ++i) a += sw2[c * 32 + i] * h[i];
      a = a > 0.f ? a : 0.f;
      u.s[j] = f2bf(a);
    }
    *(uint4*)(op + c8 * 8) = u.v4;
  }
}

// ---------------------------------------------------------------------------
// 4) generic implicit-GEMM conv via v_wmma_f32_16x16x32_bf16.
//    One wave computes a 32(oc) x 64(pos) output tile:
//      2 A fragments (weights) x 4 B fragments (im2col) -> 8 accumulators,
//      8 v_wmma per 32-wide K chunk; B fragments reused across both A rows.
//    NHWC bf16 activations; weights [OC][K] bf16, K=(kh*KS+kw)*IC+ic.
//    Epilogue: +bias, optional residual add (bf16 NHWC), optional ReLU.
// ---------------------------------------------------------------------------
__global__ __launch_bounds__(256) void conv_wmma_kernel(
    const u16t* __restrict__ in, u16t* __restrict__ out,
    const u16t* __restrict__ wq, const float* __restrict__ bias,
    const u16t* __restrict__ res,
    int NB, int H, int W, int IC, int OC, int OH, int OW,
    int stride, int pad, int KS, int doRelu) {
  const int lane = threadIdx.x & 31;
  const int wave = blockIdx.x * (blockDim.x >> 5) + (threadIdx.x >> 5);
  const int P = OH * OW;
  const int ntiles = (P + 63) >> 6;
  const int mtiles = OC >> 5;               // 32 oc per wave
  const int wpb = ntiles * mtiles;
  const int bidx = wave / wpb;
  if (bidx >= NB) return;
  const int remw = wave - bidx * wpb;
  const int mt = remw % mtiles;
  const int nt = remw / mtiles;
  const int oc0 = mt << 5;
  const int pos0 = nt << 6;
  const int n = lane & 15;  // A row (m) and B column (n)
  const int hf = lane >> 4; // K half selector
  const int Ktot = KS * KS * IC;

  int ohv[4], owv[4];
  bool pv[4];
#pragma unroll
  for (int f = 0; f < 4; ++f) {
    int p = pos0 + f * 16 + n;
    pv[f] = p < P;
    int pc = pv[f] ? p : 0;
    ohv[f] = pc / OW;
    owv[f] = pc - ohv[f] * OW;
  }

  v8f acc0[4], acc1[4];
#pragma unroll
  for (int f = 0; f < 4; ++f) { acc0[f] = fzero8(); acc1[f] = fzero8(); }

  const u16t* wr0 = wq + (size_t)(oc0 + n) * Ktot;
  const u16t* wr1 = wq + (size_t)(oc0 + 16 + n) * Ktot;
  const size_t inB = (size_t)bidx * H * W * IC;

  for (int kb = 0; kb < Ktot; kb += 32) {
    int grp = kb / IC;           // (kh*KS+kw); IC % 32 == 0 so chunks never straddle
    int ic0 = kb - grp * IC;
    int kh = grp / KS;
    int kw = grp - kh * KS;
    if (kb + 32 < Ktot) {
      __builtin_prefetch(wr0 + kb + 32, 0, 3);
      __builtin_prefetch(wr1 + kb + 32, 0, 3);
    }
    // A fragments: lane elements 0..7 <-> K kb+hf*8.., 8..15 <-> K kb+16+hf*8..
    bf16x8 a0lo = *(const bf16x8*)(wr0 + kb + hf * 8);
    bf16x8 a0hi = *(const bf16x8*)(wr0 + kb + 16 + hf * 8);
    v16bf A0 = __builtin_shufflevector(a0lo, a0hi, 0, 1, 2, 3, 4, 5, 6, 7,
                                       8, 9, 10, 11, 12, 13, 14, 15);
    bf16x8 a1lo = *(const bf16x8*)(wr1 + kb + hf * 8);
    bf16x8 a1hi = *(const bf16x8*)(wr1 + kb + 16 + hf * 8);
    v16bf A1 = __builtin_shufflevector(a1lo, a1hi, 0, 1, 2, 3, 4, 5, 6, 7,
                                       8, 9, 10, 11, 12, 13, 14, 15);
#pragma unroll
    for (int f = 0; f < 4; ++f) {
      int hin = ohv[f] * stride + kh - pad;
      int win = owv[f] * stride + kw - pad;
      bf16x8 b0 = bzero8();
      bf16x8 b1 = bzero8();
      bool ok = pv[f] & (hin >= 0) & (hin < H) & (win >= 0) & (win < W);
      if (ok) {
        const u16t* ip = in + inB + ((size_t)hin * W + win) * IC + ic0 + hf * 8;
        b0 = *(const bf16x8*)(ip);
        b1 = *(const bf16x8*)(ip + 16);
      }
      v16bf Bf = __builtin_shufflevector(b0, b1, 0, 1, 2, 3, 4, 5, 6, 7,
                                         8, 9, 10, 11, 12, 13, 14, 15);
      acc0[f] = __builtin_amdgcn_wmma_f32_16x16x32_bf16(
          false, A0, false, Bf, (short)0, acc0[f], false, false);
      acc1[f] = __builtin_amdgcn_wmma_f32_16x16x32_bf16(
          false, A1, false, Bf, (short)0, acc1[f], false, false);
    }
  }

  float bv0[8], bv1[8];
#pragma unroll
  for (int j = 0; j < 8; ++j) {
    bv0[j] = bias[oc0 + hf * 8 + j];
    bv1[j] = bias[oc0 + 16 + hf * 8 + j];
  }

#pragma unroll
  for (int f = 0; f < 4; ++f) {
    int p = pos0 + f * 16 + n;
    if (p >= P) continue;
    size_t obase = ((size_t)bidx * P + p) * OC + oc0 + hf * 8;
#pragma unroll
    for (int half2 = 0; half2 < 2; ++half2) {
      size_t ob = obase + half2 * 16;
      const v8f& a = half2 ? acc1[f] : acc0[f];
      const float* bv = half2 ? bv1 : bv0;
      union { u16t s[8]; uint4 v4; } u;
      if (res) {
        const u16t* rp = res + ob;
#pragma unroll
        for (int j = 0; j < 8; ++j) {
          float vv = a[j] + bv[j] + bf2f(rp[j]);
          if (doRelu) vv = vv > 0.f ? vv : 0.f;
          u.s[j] = f2bf(vv);
        }
      } else {
#pragma unroll
        for (int j = 0; j < 8; ++j) {
          float vv = a[j] + bv[j];
          if (doRelu) vv = vv > 0.f ? vv : 0.f;
          u.s[j] = f2bf(vv);
        }
      }
      *(uint4*)(out + ob) = u.v4;
    }
  }
}

// ---------------------------------------------------------------------------
// 5) 2x2/s2 maxpool on bf16 NHWC
// ---------------------------------------------------------------------------
__global__ __launch_bounds__(256) void pool_bf16_kernel(const u16t* __restrict__ in,
                                                        u16t* __restrict__ out,
                                                        int NB, int IH, int IW, int C) {
  int OHp = IH >> 1, OWp = IW >> 1;
  size_t total = (size_t)NB * OHp * OWp * C;
  size_t idx = (size_t)blockIdx.x * blockDim.x + threadIdx.x;
  if (idx >= total) return;
  int c = (int)(idx % C);
  size_t t = idx / C;
  int ow = (int)(t % OWp);
  t /= OWp;
  int oh = (int)(t % OHp);
  int b = (int)(t / OHp);
  size_t base = (((size_t)b * IH + 2 * oh) * IW + 2 * ow) * C + c;
  float m0 = bf2f(in[base]);
  float m1 = bf2f(in[base + C]);
  float m2 = bf2f(in[base + (size_t)IW * C]);
  float m3 = bf2f(in[base + (size_t)IW * C + C]);
  float m = fmaxf(fmaxf(m0, m1), fmaxf(m2, m3));
  out[idx] = f2bf(m);
}

// ---------------------------------------------------------------------------
// 6) final 2x2/s2 maxpool: bf16 NHWC -> f32 NCHW (matches reference output)
// ---------------------------------------------------------------------------
__global__ __launch_bounds__(256) void pool_final_kernel(const u16t* __restrict__ in,
                                                         float* __restrict__ out,
                                                         int NB, int IH, int IW, int C) {
  int OWp = IW >> 1;
  size_t total = (size_t)NB * C * OWp;
  size_t idx = (size_t)blockIdx.x * blockDim.x + threadIdx.x;
  if (idx >= total) return;
  int ow = (int)(idx % OWp);
  size_t t = idx / OWp;
  int c = (int)(t % C);
  int b = (int)(t / C);
  float m = -__builtin_inff();
#pragma unroll
  for (int ih = 0; ih < 2; ++ih)
#pragma unroll
    for (int dw = 0; dw < 2; ++dw) {
      size_t ii = (((size_t)b * IH + ih) * IW + 2 * ow + dw) * C + c;
      m = fmaxf(m, bf2f(in[ii]));
    }
  out[idx] = m;  // NCHW flat: ((b*C+c)*1 + 0)*OW + ow
}

// ---------------------------------------------------------------------------
extern "C" void kernel_launch(void* const* d_in, const int* in_sizes, int n_in,
                              void* d_out, int out_size, void* d_ws, size_t ws_size,
                              hipStream_t stream) {
  (void)in_sizes; (void)n_in; (void)out_size; (void)ws_size;
  const float* pts = (const float*)d_in[0];
#define FP(i) ((const float*)d_in[i])

  char* ws = (char*)d_ws;
  size_t off = 0;
  auto alloc = [&](size_t bytes) -> char* {
    char* p = ws + off;
    off += (bytes + 255) & ~(size_t)255;
    return p;
  };

  // folded FE params
  float* fw1 = (float*)alloc(96 * 4);
  float* fb1 = (float*)alloc(32 * 4);
  float* fw2 = (float*)alloc(2048 * 4);
  float* fb2 = (float*)alloc(64 * 4);
  // quantized conv weights + folded biases
  u16t* wq_c0 = (u16t*)alloc(64 * 576 * 2);      float* bq_c0 = (float*)alloc(64 * 4);
  u16t* wq_r1c1 = (u16t*)alloc(128 * 576 * 2);   float* bq_r1c1 = (float*)alloc(128 * 4);
  u16t* wq_r1c2 = (u16t*)alloc(128 * 1152 * 2);  float* bq_r1c2 = (float*)alloc(128 * 4);
  u16t* wq_r1sc = (u16t*)alloc(128 * 64 * 2);    float* bq_r1sc = (float*)alloc(128 * 4);
  u16t* wq_r2c1 = (u16t*)alloc(256 * 1152 * 2);  float* bq_r2c1 = (float*)alloc(256 * 4);
  u16t* wq_r2c2 = (u16t*)alloc(256 * 2304 * 2);  float* bq_r2c2 = (float*)alloc(256 * 4);
  u16t* wq_r2sc = (u16t*)alloc(256 * 128 * 2);   float* bq_r2sc = (float*)alloc(256 * 4);

  int* last = (int*)alloc((size_t)NBATCH * NVc * 4);                    // 2.82 MB
  u16t* img = (u16t*)alloc((size_t)NBATCH * NVc * 64 * 2);              // 90.1 MB
  u16t* y0 = (u16t*)alloc((size_t)NBATCH * 88000 * 64 * 2);             // 22.5 MB
  u16t* r1o = (u16t*)alloc((size_t)NBATCH * 88000 * 128 * 2);           // 45.1 MB
  u16t* p1 = (u16t*)alloc((size_t)NBATCH * 17600 * 128 * 2);            // 9.0 MB
  // aliases (lifetimes disjoint on the in-order stream):
  u16t* t1 = img;                                     // c1 out, after img consumed by c0
  u16t* s1 = img + (size_t)NBATCH * 88000 * 128;      // sc out (img is exactly t1+s1 sized)
  u16t* t2 = r1o;                                     // r2 c1 out, after pool1 read r1o
  u16t* s2 = r1o + (size_t)NBATCH * 17600 * 256;      // r2 sc out
  u16t* r2o = y0;                                     // r2 out, after y0 consumed by r1

  // ---- 1) init + scatter ----
  (void)hipMemsetAsync(last, 0xFF, (size_t)NBATCH * NVc * 4, stream);  // -1
  scatter_kernel<<<(NBATCH * NPOINTS + 255) / 256, 256, 0, stream>>>(pts, last);

  // ---- 2) param prep ----
  prep_fe_kernel<<<1, 256, 0, stream>>>(FP(1), FP(2), FP(3), FP(4), FP(5), FP(6),
                                        FP(7), FP(8), FP(9), FP(10), FP(11), FP(12),
                                        fw1, fb1, fw2, fb2);
  auto prep = [&](int wi, u16t* wq, float* bq, int OC, int IC, int KS) {
    int total = OC * KS * KS * IC;
    prep_conv_kernel<<<(total + 255) / 256, 256, 0, stream>>>(
        FP(wi), FP(wi + 1), FP(wi + 2), FP(wi + 3), FP(wi + 4), FP(wi + 5),
        wq, bq, OC, IC, KS);
  };
  prep(13, wq_c0, bq_c0, 64, 64, 3);
  prep(19, wq_r1c1, bq_r1c1, 128, 64, 3);
  prep(25, wq_r1c2, bq_r1c2, 128, 128, 3);
  prep(31, wq_r1sc, bq_r1sc, 128, 64, 1);
  prep(37, wq_r2c1, bq_r2c1, 256, 128, 3);
  prep(43, wq_r2c2, bq_r2c2, 256, 256, 3);
  prep(49, wq_r2sc, bq_r2sc, 256, 128, 1);

  // ---- 3) feature extractor -> bf16 NHWC pseudo-image [2][10][35200][64] ----
  fe_kernel<<<(NBATCH * NVc + 255) / 256, 256, 0, stream>>>(last, pts, fw1, fb1,
                                                            fw2, fb2, img);

  // ---- 4) WMMA conv backbone ----
  auto conv = [&](const u16t* in, u16t* out, const u16t* wq, const float* bq,
                  const u16t* res, int H, int W, int IC, int OC, int OH, int OW,
                  int st, int pd, int ks, int relu) {
    int P = OH * OW;
    int waves = NBATCH * ((P + 63) / 64) * (OC / 32);
    int blocks = (waves + 7) / 8;
    conv_wmma_kernel<<<blocks, 256, 0, stream>>>(in, out, wq, bq, res, NBATCH, H, W,
                                                 IC, OC, OH, OW, st, pd, ks, relu);
  };

  // c0: [10,35200,64] -s2-> [5,17600,64], ReLU
  conv(img, y0, wq_c0, bq_c0, nullptr, 10, 35200, 64, 64, 5, 17600, 2, 1, 3, 1);
  // r1 shortcut 1x1 64->128 (no relu)
  conv(y0, s1, wq_r1sc, bq_r1sc, nullptr, 5, 17600, 64, 128, 5, 17600, 1, 0, 1, 0);
  // r1 c1 3x3 64->128, ReLU
  conv(y0, t1, wq_r1c1, bq_r1c1, nullptr, 5, 17600, 64, 128, 5, 17600, 1, 1, 3, 1);
  // r1 c2 3x3 128->128 + residual(s1), ReLU
  conv(t1, r1o, wq_r1c2, bq_r1c2, s1, 5, 17600, 128, 128, 5, 17600, 1, 1, 3, 1);
  // pool1: [5,17600,128] -> [2,8800,128]
  {
    size_t total = (size_t)NBATCH * 2 * 8800 * 128;
    pool_bf16_kernel<<<(int)((total + 255) / 256), 256, 0, stream>>>(r1o, p1, NBATCH,
                                                                     5, 17600, 128);
  }
  // r2 shortcut 1x1 128->256
  conv(p1, s2, wq_r2sc, bq_r2sc, nullptr, 2, 8800, 128, 256, 2, 8800, 1, 0, 1, 0);
  // r2 c1 3x3 128->256, ReLU
  conv(p1, t2, wq_r2c1, bq_r2c1, nullptr, 2, 8800, 128, 256, 2, 8800, 1, 1, 3, 1);
  // r2 c2 3x3 256->256 + residual(s2), ReLU
  conv(t2, r2o, wq_r2c2, bq_r2c2, s2, 2, 8800, 256, 256, 2, 8800, 1, 1, 3, 1);
  // final pool -> f32 NCHW [2][256][1][4400]
  {
    size_t total = (size_t)NBATCH * 256 * 4400;
    pool_final_kernel<<<(int)((total + 255) / 256), 256, 0, stream>>>(
        r2o, (float*)d_out, NBATCH, 2, 8800, 256);
  }
}
